// RPN_26371099197850
// MI455X (gfx1250) — matrix operand order repaired and checked by reading, not compiled, not run on hardware
//
#include <hip/hip_runtime.h>
#include <hip/hip_bf16.h>

// ---------------------------------------------------------------------------
// RPN loss for MI455X (gfx1250): bf16 WMMA conv3x3 + fused conv1x1, anchor
// labeling, masked CE reduction. A-panels staged to LDS via CDNA5 async
// global->LDS loads (double buffered, s_wait_asynccnt). wave32 layouts per
// CDNA5 ISA.
// ---------------------------------------------------------------------------

typedef __attribute__((ext_vector_type(16))) __bf16 v16bf;
typedef __attribute__((ext_vector_type(8)))  __bf16 v8bf;
typedef __attribute__((ext_vector_type(8)))  float  v8f;

#define FH     128
#define FW     128
#define CIN    512
#define CMID   256
#define KANCH  9
#define NPOS   (FH * FW)            // 16384
#define NA     (NPOS * KANCH)       // 147456
#define PADW   130
#define NGT    64
#define IMGSZ  2048.0f
#define NLBLK  576                  // NA / 256
#define MTILE  32                   // positions per workgroup (one row segment)

// float -> bf16 (round to nearest even)
__device__ inline unsigned short f2bf(float f) {
  unsigned int u = __float_as_uint(f);
  unsigned int r = (u + 0x7FFFu + ((u >> 16) & 1u)) >> 16;
  return (unsigned short)r;
}

__device__ inline void anchor_coords(int a, float& x1, float& y1,
                                     float& x2, float& y2) {
  int k   = a % KANCH;
  int pos = a / KANCH;
  int w   = pos % FW;
  int h   = pos / FW;
  const float ratios[3] = {0.5f, 1.0f, 2.0f};
  const float scales[3] = {8.0f, 16.0f, 32.0f};
  float r  = ratios[k / 3];
  float s  = scales[k % 3];
  float ws = roundf(sqrtf(256.0f / r));
  float hs = roundf(ws * r);
  float bw = ws * s, bh = hs * s;
  float sx = (float)w * 16.0f, sy = (float)h * 16.0f;
  x1 = sx + 7.5f - (bw - 1.0f) * 0.5f;
  y1 = sy + 7.5f - (bh - 1.0f) * 0.5f;
  x2 = sx + 7.5f + (bw - 1.0f) * 0.5f;
  y2 = sy + 7.5f + (bh - 1.0f) * 0.5f;
}

__device__ inline float iou_one(float ax1, float ay1, float ax2, float ay2,
                                float gx1, float gy1, float gx2, float gy2) {
  float areaA = (ax2 - ax1 + 1.0f) * (ay2 - ay1 + 1.0f);
  float areaG = (gx2 - gx1 + 1.0f) * (gy2 - gy1 + 1.0f);
  float ix1 = fmaxf(ax1, gx1), iy1 = fmaxf(ay1, gy1);
  float ix2 = fminf(ax2, gx2), iy2 = fminf(ay2, gy2);
  float iw = fmaxf(ix2 - ix1 + 1.0f, 0.0f);
  float ih = fmaxf(iy2 - iy1 + 1.0f, 0.0f);
  float inter = iw * ih;
  return inter / (areaA + areaG - inter);
}

// --- 1) pad + convert x (f32 NHWC 128x128x512) -> bf16 130x130x512 ----------
__global__ __launch_bounds__(256) void pad_convert_kernel(
    const float* __restrict__ x, unsigned short* __restrict__ xpad) {
  int idx = blockIdx.x * 256 + threadIdx.x;       // < 130*130*512
  int c  = idx & (CIN - 1);
  int p  = idx >> 9;
  int px = p % PADW, py = p / PADW;
  float v = 0.0f;
  if (py >= 1 && py <= FH && px >= 1 && px <= FW)
    v = x[((size_t)((py - 1) * FW + (px - 1))) * CIN + c];
  xpad[idx] = f2bf(v);
}

// --- 2a) pack shared_W [3,3,512,256] into WMMA-B lane records ---------------
// record r = ((tap*16 + kc)*16 + ntile)*32 + lane; 16 bf16 per record:
//   lane<16  -> column n = ntile*16+lane,    K = kc*32 + j        (j=0..15)
//   lane>=16 -> column n = ntile*16+lane-16, K = kc*32 + 16 + j
__global__ __launch_bounds__(256) void pack_sharedW_kernel(
    const float* __restrict__ W, unsigned short* __restrict__ wpk) {
  int r = blockIdx.x * 256 + threadIdx.x;         // < 9*16*16*32 = 73728
  int lane = r & 31;
  int t  = r >> 5;
  int ntile = t & 15;
  int t2 = t >> 4;
  int kc  = t2 & 15;
  int tap = t2 >> 4;
  int n     = ntile * 16 + (lane & 15);
  int kbase = kc * 32 + ((lane >> 4) ? 16 : 0);
  unsigned short* dst = wpk + (size_t)r * 16;
#pragma unroll
  for (int j = 0; j < 16; ++j)
    dst[j] = f2bf(W[((size_t)(tap * CIN + kbase + j)) * CMID + n]);
}

// --- 2b) pack cls_W [256,18] into WMMA-B records, N padded to 32 ------------
__global__ __launch_bounds__(256) void pack_clsW_kernel(
    const float* __restrict__ W, unsigned short* __restrict__ cpk) {
  int r = blockIdx.x * 256 + threadIdx.x;         // < 8*2*32 = 512
  if (r >= 512) return;
  int lane = r & 31;
  int t  = r >> 5;
  int nt = t & 1;
  int kc = t >> 1;
  int n     = nt * 16 + (lane & 15);
  int kbase = kc * 32 + ((lane >> 4) ? 16 : 0);
  unsigned short* dst = cpk + (size_t)r * 16;
#pragma unroll
  for (int j = 0; j < 16; ++j)
    dst[j] = (n < 2 * KANCH) ? f2bf(W[(size_t)(kbase + j) * (2 * KANCH) + n])
                             : (unsigned short)0;
}

// --- 3) per-gt max IoU over all anchors (gt_max) ----------------------------
__global__ __launch_bounds__(256) void gtmax_kernel(
    const float* __restrict__ gt, float* __restrict__ gt_max) {
  __shared__ float red[256];
  int g = blockIdx.x;
  float gx1 = gt[g * 4 + 0], gy1 = gt[g * 4 + 1];
  float gx2 = gt[g * 4 + 2], gy2 = gt[g * 4 + 3];
  float mx = -1.0f;
  for (int a = threadIdx.x; a < NA; a += 256) {
    float x1, y1, x2, y2;
    anchor_coords(a, x1, y1, x2, y2);
    mx = fmaxf(mx, iou_one(x1, y1, x2, y2, gx1, gy1, gx2, gy2));
  }
  red[threadIdx.x] = mx;
  __syncthreads();
  for (int off = 128; off; off >>= 1) {
    if (threadIdx.x < off)
      red[threadIdx.x] = fmaxf(red[threadIdx.x], red[threadIdx.x + off]);
    __syncthreads();
  }
  if (threadIdx.x == 0) gt_max[g] = red[0];
}

// --- 4) per-anchor fg/bg codes: 1=fg, 2=bg, 0=ignore ------------------------
__global__ __launch_bounds__(256) void codes_kernel(
    const float* __restrict__ gt, const float* __restrict__ gt_max,
    int* __restrict__ codes) {
  int a = blockIdx.x * 256 + threadIdx.x;         // exact: 576*256 = NA
  float x1, y1, x2, y2;
  anchor_coords(a, x1, y1, x2, y2);
  bool inside = (x1 >= 0.0f) && (y1 >= 0.0f) && (x2 < IMGSZ) && (y2 < IMGSZ);
  float maxov = 0.0f;
  bool best = false;
  for (int g = 0; g < NGT; ++g) {
    float i = iou_one(x1, y1, x2, y2, gt[g * 4 + 0], gt[g * 4 + 1],
                      gt[g * 4 + 2], gt[g * 4 + 3]);
    maxov = fmaxf(maxov, i);
    best = best || (i == gt_max[g]);
  }
  bool fg = inside && (best || maxov >= 0.7f);
  bool bg = inside && (maxov < 0.3f) && !fg;
  codes[a] = fg ? 1 : (bg ? 2 : 0);
}

// --- 5) deterministic label selection (single workgroup, prefix scan) -------
__global__ __launch_bounds__(1024) void select_labels_kernel(
    const int* __restrict__ codes, int* __restrict__ labels) {
  __shared__ int sfg[1024];
  __shared__ int sbg[1024];
  __shared__ int base_fg, base_bg, s_numfg, s_bglim;
  const int tid = threadIdx.x;

  // pass 1: total fg count
  int cnt = 0;
  for (int a = tid; a < NA; a += 1024) cnt += (codes[a] == 1);
  sfg[tid] = cnt;
  __syncthreads();
  for (int off = 512; off; off >>= 1) {
    if (tid < off) sfg[tid] += sfg[tid + off];
    __syncthreads();
  }
  if (tid == 0) {
    int fg_total = sfg[0];
    int num_fg = fg_total < 128 ? fg_total : 128;
    s_numfg = num_fg;
    s_bglim = 256 - num_fg;
    base_fg = 0;
    base_bg = 0;
  }
  __syncthreads();
  const int num_fg = s_numfg, bg_lim = s_bglim;

  // pass 2: sequential chunked inclusive scan, keep first num_fg / bg_lim
  for (int chunk = 0; chunk < NA; chunk += 1024) {
    int a = chunk + tid;
    int code = (a < NA) ? codes[a] : 0;
    int f = (code == 1) ? 1 : 0;
    int b = (code == 2) ? 1 : 0;
    sfg[tid] = f;
    sbg[tid] = b;
    __syncthreads();
    for (int off = 1; off < 1024; off <<= 1) {
      int vf = (tid >= off) ? sfg[tid - off] : 0;
      int vb = (tid >= off) ? sbg[tid - off] : 0;
      __syncthreads();
      sfg[tid] += vf;
      sbg[tid] += vb;
      __syncthreads();
    }
    int bf = base_fg, bb = base_bg;
    int lab = -1;
    if (f && (bf + sfg[tid] - 1) < num_fg) lab = 1;
    else if (b && (bb + sbg[tid] - 1) < bg_lim) lab = 0;
    if (a < NA) labels[a] = lab;
    __syncthreads();
    if (tid == 0) {
      base_fg = bf + sfg[1023];
      base_bg = bb + sbg[1023];
    }
    __syncthreads();
  }
}

// --- 6) main WMMA kernel: conv3x3 (bf16 WMMA, f32 acc) + fused conv1x1 ------
// grid: 512 M-tiles of 32 positions (one row segment, no row wrap).
// block: 256 threads = 8 wave32; wave w owns N-tiles {2w,2w+1} x both M-tiles.
// A panel per tap (32 pos x 512 ch bf16 = 32 KB) staged in LDS via CDNA5
// async global->LDS b128 loads, double-buffered across taps (ASYNCcnt).
__global__ __launch_bounds__(256) void conv_wmma_kernel(
    const unsigned short* __restrict__ xpad,
    const unsigned short* __restrict__ wpk,
    const unsigned short* __restrict__ cpk,
    const float* __restrict__ shared_b,
    const float* __restrict__ cls_b,
    float* __restrict__ logits) {
  __shared__ unsigned short shmem[2 * MTILE * CIN];  // 64 KB A double buffer
  unsigned short* shC = shmem;  // cls staging (16 KB) aliases buf0 post-conv

  const int tid  = threadIdx.x;
  const int lane = tid & 31;
  const int wave = tid >> 5;
  const int mt = blockIdx.x;                 // 0..511
  const int h  = mt >> 2;
  const int w0 = (mt & 3) << 5;              // 32-position row segment
  const int m    = lane & 15;                // A row / D column index
  const int hi16 = lane >> 4;                // upper half-wave flag
  const int koffA = hi16 ? 16 : 0;           // A K-layout: contiguous 16 per half

  // async copy mapping: thread t owns 128 B of the panel (8 x b128)
  const int arow = tid >> 3;                 // 0..31 (position)
  const int acol = (tid & 7) * 64;           // ushort offset in channel dim

  auto issue_panel = [&](int tap, int buf) {
    int dy = tap / 3 - 1, dx = tap % 3 - 1;
    int py = h + dy + 1;
    int px = w0 + dx + 1 + arow;
    const unsigned short* g = xpad + ((size_t)(py * PADW + px)) * CIN + acol;
    unsigned lds =
        (unsigned)(size_t)(shmem + buf * MTILE * CIN + arow * CIN + acol);
#pragma unroll
    for (int i = 0; i < 8; ++i) {
      asm volatile("global_load_async_to_lds_b128 %0, %1, off"
                   :: "v"(lds + i * 16),
                      "v"((unsigned long long)(size_t)(g + i * 8))
                   : "memory");
    }
  };

  v8f acc[2][2] = {};                        // [mtile][ntile]

  issue_panel(0, 0);
  for (int tap = 0; tap < 9; ++tap) {
    const int b = tap & 1;
    if (tap < 8) {
      issue_panel(tap + 1, b ^ 1);
      asm volatile("s_wait_asynccnt 0x8" ::: "memory");  // current panel done
    } else {
      asm volatile("s_wait_asynccnt 0x0" ::: "memory");
    }
    __syncthreads();
    const unsigned short* pa = shmem + b * MTILE * CIN;
    for (int kc = 0; kc < 16; ++kc) {
      const unsigned short* bp =
          wpk + ((size_t)(((tap * 16 + kc) * 16 + wave * 2) * 32 + lane)) * 16;
      v16bf B0 = *reinterpret_cast<const v16bf*>(bp);
      v16bf B1 = *reinterpret_cast<const v16bf*>(bp + 32 * 16);
#pragma unroll
      for (int mt2 = 0; mt2 < 2; ++mt2) {
        const unsigned short* ap = pa + (mt2 * 16 + m) * CIN + kc * 32 + koffA;
        v8bf alo = *reinterpret_cast<const v8bf*>(ap);
        v8bf ahi = *reinterpret_cast<const v8bf*>(ap + 8);
        v16bf Av = __builtin_shufflevector(alo, ahi, 0, 1, 2, 3, 4, 5, 6, 7,
                                           8, 9, 10, 11, 12, 13, 14, 15);
        acc[mt2][0] = __builtin_amdgcn_wmma_f32_16x16x32_bf16(
            false, Av, false, B0, (short)0, acc[mt2][0], false, false);
        acc[mt2][1] = __builtin_amdgcn_wmma_f32_16x16x32_bf16(
            false, Av, false, B1, (short)0, acc[mt2][1], false, false);
      }
    }
    __syncthreads();                         // panel consumed; safe to overwrite
  }

  // add shared bias, stage D (M=pos, N=chan) to LDS as bf16 (32 x 256)
  {
    int ch0 = (wave * 2 + 0) * 16 + m;
    int ch1 = (wave * 2 + 1) * 16 + m;
    float b0 = shared_b[ch0], b1 = shared_b[ch1];
#pragma unroll
    for (int mt2 = 0; mt2 < 2; ++mt2)
#pragma unroll
      for (int j = 0; j < 8; ++j) {
        int row = mt2 * 16 + j + hi16 * 8;   // D row layout (ISA table)
        shC[row * CMID + ch0] = f2bf(acc[mt2][0][j] + b0);
        shC[row * CMID + ch1] = f2bf(acc[mt2][1][j] + b1);
      }
  }
  __syncthreads();

  // fused 1x1 cls conv: waves 0,1 each take one 16-pos M-tile.
  if (wave < 2) {
    v8f c0 = {};
    v8f c1 = {};
#pragma unroll
    for (int kc = 0; kc < 8; ++kc) {
      const unsigned short* ap =
          &shC[(wave * 16 + m) * CMID + kc * 32 + koffA];
      v8bf alo = *reinterpret_cast<const v8bf*>(ap);
      v8bf ahi = *reinterpret_cast<const v8bf*>(ap + 8);
      v16bf Av = __builtin_shufflevector(alo, ahi, 0, 1, 2, 3, 4, 5, 6, 7, 8,
                                         9, 10, 11, 12, 13, 14, 15);
      const unsigned short* bp = cpk + ((size_t)((kc * 2) * 32 + lane)) * 16;
      v16bf B0 = *reinterpret_cast<const v16bf*>(bp);
      v16bf B1 = *reinterpret_cast<const v16bf*>(bp + 32 * 16);
      c0 = __builtin_amdgcn_wmma_f32_16x16x32_bf16(false, Av, false, B0,
                                                   (short)0, c0, false, false);
      c1 = __builtin_amdgcn_wmma_f32_16x16x32_bf16(false, Av, false, B1,
                                                   (short)0, c1, false, false);
    }
#pragma unroll
    for (int j = 0; j < 8; ++j) {
      int row  = j + hi16 * 8;
      int gpos = mt * MTILE + wave * 16 + row;  // == h*128 + w0 + ...
      int n0 = m;                               // 0..15, always < 18
      logits[(size_t)gpos * (2 * KANCH) + n0] = c0[j] + cls_b[n0];
      int n1 = 16 + m;
      if (n1 < 2 * KANCH)
        logits[(size_t)gpos * (2 * KANCH) + n1] = c1[j] + cls_b[n1];
    }
  }
}

// --- 7) masked CE partial reduction ----------------------------------------
__global__ __launch_bounds__(256) void loss_partial_kernel(
    const float* __restrict__ logits, const int* __restrict__ labels,
    float* __restrict__ psum, int* __restrict__ pcnt) {
  __shared__ float rs[256];
  __shared__ int   rc[256];
  const int tid = threadIdx.x;
  int a = blockIdx.x * 256 + tid;                 // exact: 576*256 = NA
  int lab = labels[a];
  float ce = 0.0f;
  int c = 0;
  if (lab >= 0) {
    int pos = a / KANCH, k = a - pos * KANCH;
    float lb = logits[(size_t)pos * (2 * KANCH) + 2 * k + 0];
    float lf = logits[(size_t)pos * (2 * KANCH) + 2 * k + 1];
    float mx  = fmaxf(lb, lf);
    float lse = mx + logf(__expf(lb - mx) + __expf(lf - mx));
    ce = lse - (lab ? lf : lb);
    c = 1;
  }
  rs[tid] = ce;
  rc[tid] = c;
  __syncthreads();
  for (int off = 128; off; off >>= 1) {
    if (tid < off) {
      rs[tid] += rs[tid + off];
      rc[tid] += rc[tid + off];
    }
    __syncthreads();
  }
  if (tid == 0) {
    psum[blockIdx.x] = rs[0];
    pcnt[blockIdx.x] = rc[0];
  }
}

// --- 8) deterministic final reduce -----------------------------------------
__global__ void finalize_kernel(const float* __restrict__ psum,
                                const int* __restrict__ pcnt,
                                float* __restrict__ out) {
  if (threadIdx.x == 0 && blockIdx.x == 0) {
    float s = 0.0f;
    int c = 0;
    for (int i = 0; i < NLBLK; ++i) {
      s += psum[i];
      c += pcnt[i];
    }
    out[0] = s / (float)(c > 1 ? c : 1);
  }
}

// ---------------------------------------------------------------------------
extern "C" void kernel_launch(void* const* d_in, const int* in_sizes, int n_in,
                              void* d_out, int out_size, void* d_ws,
                              size_t ws_size, hipStream_t stream) {
  (void)in_sizes; (void)n_in; (void)out_size; (void)ws_size;
  const float* x        = (const float*)d_in[0];
  const float* shared_W = (const float*)d_in[1];
  const float* shared_b = (const float*)d_in[2];
  const float* cls_W    = (const float*)d_in[3];
  const float* cls_b    = (const float*)d_in[4];
  // d_in[5], d_in[6]: bbox_W/b — unused by the loss, not computed.
  const float* gt_boxes = (const float*)d_in[7];
  float* out = (float*)d_out;

  // workspace carve (all 256B aligned)
  char* ws = (char*)d_ws;
  size_t o = 0;
  auto take = [&](size_t bytes) {
    char* p = ws + o;
    o = (o + bytes + 255) & ~(size_t)255;
    return p;
  };
  unsigned short* xpad   = (unsigned short*)take((size_t)PADW * PADW * CIN * 2);
  unsigned short* wpk    = (unsigned short*)take((size_t)9 * 16 * 16 * 32 * 16 * 2);
  unsigned short* cpk    = (unsigned short*)take((size_t)8 * 2 * 32 * 16 * 2);
  float*          logits = (float*)take((size_t)NPOS * 2 * KANCH * 4);
  float*          gt_max = (float*)take((size_t)NGT * 4);
  int*            codes  = (int*)take((size_t)NA * 4);
  int*            labels = (int*)take((size_t)NA * 4);
  float*          psum   = (float*)take((size_t)NLBLK * 4);
  int*            pcnt   = (int*)take((size_t)NLBLK * 4);

  // 1) data prep (bf16 conversion + WMMA-layout packing)
  pad_convert_kernel<<<(PADW * PADW * CIN) / 256, 256, 0, stream>>>(x, xpad);
  pack_sharedW_kernel<<<(9 * 16 * 16 * 32) / 256, 256, 0, stream>>>(shared_W, wpk);
  pack_clsW_kernel<<<2, 256, 0, stream>>>(cls_W, cpk);

  // 2) anchor labeling
  gtmax_kernel<<<NGT, 256, 0, stream>>>(gt_boxes, gt_max);
  codes_kernel<<<NA / 256, 256, 0, stream>>>(gt_boxes, gt_max, codes);
  select_labels_kernel<<<1, 1024, 0, stream>>>(codes, labels);

  // 3) WMMA conv3x3 + fused conv1x1 -> per-anchor logits
  conv_wmma_kernel<<<NPOS / MTILE, 256, 0, stream>>>(xpad, wpk, cpk, shared_b,
                                                     cls_b, logits);

  // 4) masked CE loss
  loss_partial_kernel<<<NLBLK, 256, 0, stream>>>(logits, labels, psum, pcnt);
  finalize_kernel<<<1, 64, 0, stream>>>(psum, pcnt, out);
}
